// OADModel_48249662603572
// MI455X (gfx1250) — matrix operand order repaired and checked by reading, not compiled, not run on hardware
//
#include <hip/hip_runtime.h>
#include <hip/hip_bf16.h>

typedef __bf16 bf16;
typedef __attribute__((ext_vector_type(16))) __bf16 v16bf;
typedef __attribute__((ext_vector_type(8)))  float  v8f;

#define BATCH 64
#define SEQ   512
#define FEAT  3072
#define HID   1024
#define GATES 4096
#define NCLS  22
#define MROWS (BATCH*SEQ)

#define WMMA_BF16(a,b,c) \
  __builtin_amdgcn_wmma_f32_16x16x32_bf16(false,(a),false,(b),(short)0,(c),false,false)

union FragU { uint4 u[2]; v16bf v; };

__device__ __forceinline__ float sigf(float x) { return 1.0f / (1.0f + __expf(-x)); }

// Low 32 bits of the flat address of an LDS object = its LDS byte offset
// (LDS aperture tag lives in bits [63:32]).
__device__ __forceinline__ unsigned lds_off32(const void* p) {
  return (unsigned)(unsigned long long)(uintptr_t)p;
}

// Async copy 32 bytes global -> LDS (two b128 transfers, ASYNCcnt-tracked).
// INST_OFFSET applies to both the LDS and global addresses per the ISA.
__device__ __forceinline__ void async_ld_b256(unsigned lds_addr, const bf16* gp) {
  asm volatile("global_load_async_to_lds_b128 %0, %1, off\n\t"
               "global_load_async_to_lds_b128 %0, %1, off offset:16"
               :: "v"(lds_addr), "v"(gp)
               : "memory");
}

__device__ __forceinline__ void wait_async0() {
  asm volatile("s_wait_asynccnt 0x0" ::: "memory");
}

// ---------------------------------------------------------------------------
// Generic bf16 WMMA GEMM: C[M,N] = act(A[M,K] * B[N,K]^T + bias)
// AMODE: 0 = A fp32 (converted to bf16 manually), 1 = A bf16 (async->LDS)
// OMODE: 0 = store fp32, 1 = relu + store bf16
// Block tile 128x128, 8 wave32s (4 M x 2 N), K-chunk 32, double-buffered LDS
// filled with GLOBAL_LOAD_ASYNC_TO_LDS_B128 so HBM/L2 traffic overlaps WMMA.
// ---------------------------------------------------------------------------
template<int AMODE, int OMODE>
__global__ __launch_bounds__(256) void gemm_bf16_wmma(
    const void* __restrict__ Aptr, const bf16* __restrict__ Bw,
    const float* __restrict__ bias, void* __restrict__ Cptr,
    int M, int N, int K, int ldc)
{
  __shared__ __align__(16) bf16 As[2][128 * 32];
  __shared__ __align__(16) bf16 Bs[2][128 * 32];

  const int tid  = threadIdx.x;
  const int lane = tid & 31, wid = tid >> 5;
  const int wm = wid >> 1, wn = wid & 1;
  const int bm = blockIdx.x * 128, bn = blockIdx.y * 128;
  const int r   = lane & 15;
  const int koA = (lane >> 4) * 8;    // A frag: two 8-elem chunks at ko, ko+16
  const int koB = (lane >> 4) * 8;    // B frag: two 8-elem chunks at ko, ko+16

  const int rowF = tid >> 1;          // 0..127
  const int colF = (tid & 1) * 16;    // 0 or 16
  const int gnF  = bn + rowF;

  // Zero-pad B rows >= N once; async fills never touch them (EXEC-masked).
  if (gnF >= N) {
    uint4 z; z.x = z.y = z.z = z.w = 0u;
#pragma unroll
    for (int b = 0; b < 2; b++) {
      uint4* d = (uint4*)&Bs[b][rowF * 32 + colF];
      d[0] = z; d[1] = z;
    }
  }

  v8f zero = {};
  v8f acc[2][4];
#pragma unroll
  for (int i = 0; i < 2; i++)
#pragma unroll
    for (int j = 0; j < 4; j++) acc[i][j] = zero;

  auto fill = [&](int buf, int k0) {
    if (AMODE == 0) {
      // fp32 A: load, convert to bf16, store to LDS manually.
      const float* Ap = (const float*)Aptr + (size_t)(bm + rowF) * K + k0 + colF;
      float4 f0 = *(const float4*)(Ap + 0);
      float4 f1 = *(const float4*)(Ap + 4);
      float4 f2 = *(const float4*)(Ap + 8);
      float4 f3 = *(const float4*)(Ap + 12);
      union { bf16 h[16]; uint4 u[2]; } cv;
      cv.h[0]=(bf16)f0.x; cv.h[1]=(bf16)f0.y; cv.h[2]=(bf16)f0.z; cv.h[3]=(bf16)f0.w;
      cv.h[4]=(bf16)f1.x; cv.h[5]=(bf16)f1.y; cv.h[6]=(bf16)f1.z; cv.h[7]=(bf16)f1.w;
      cv.h[8]=(bf16)f2.x; cv.h[9]=(bf16)f2.y; cv.h[10]=(bf16)f2.z; cv.h[11]=(bf16)f2.w;
      cv.h[12]=(bf16)f3.x; cv.h[13]=(bf16)f3.y; cv.h[14]=(bf16)f3.z; cv.h[15]=(bf16)f3.w;
      uint4* d = (uint4*)&As[buf][rowF * 32 + colF];
      d[0] = cv.u[0]; d[1] = cv.u[1];
    } else {
      // bf16 A: async DMA straight into LDS.
      const bf16* Ap = (const bf16*)Aptr + (size_t)(bm + rowF) * K + k0 + colF;
      async_ld_b256(lds_off32(&As[buf][rowF * 32 + colF]), Ap);
    }
    if (gnF < N) {
      const bf16* Bp = Bw + (size_t)gnF * K + k0 + colF;
      async_ld_b256(lds_off32(&Bs[buf][rowF * 32 + colF]), Bp);
    }
  };

  fill(0, 0);
  const int niter = K / 32;
  for (int it = 0; it < niter; ++it) {
    const int cur = it & 1;
    wait_async0();        // my async fills for buf `cur` have landed
    __syncthreads();      // everyone's fills landed; prev reads of buf cur^1 done
    if (it + 1 < niter) fill(cur ^ 1, (it + 1) * 32);

    FragU af[2], bfb[4];
#pragma unroll
    for (int i = 0; i < 2; i++) {
      const uint4* p = (const uint4*)&As[cur][(wm * 32 + i * 16 + r) * 32 + koA];
      af[i].u[0] = p[0]; af[i].u[1] = p[2];   // +16 elements
    }
#pragma unroll
    for (int j = 0; j < 4; j++) {
      const uint4* p = (const uint4*)&Bs[cur][(wn * 64 + j * 16 + r) * 32 + koB];
      bfb[j].u[0] = p[0]; bfb[j].u[1] = p[2];
    }
#pragma unroll
    for (int i = 0; i < 2; i++)
#pragma unroll
      for (int j = 0; j < 4; j++)
        acc[i][j] = WMMA_BF16(af[i].v, bfb[j].v, acc[i][j]);
  }

  // ---- epilogue: C layout row = e + 8*(lane>>4), col = lane&15 ----
  const int rb = (lane >> 4) * 8;
  const int cc = lane & 15;
#pragma unroll
  for (int i = 0; i < 2; i++) {
#pragma unroll
    for (int j = 0; j < 4; j++) {
      const int gn = bn + wn * 64 + j * 16 + cc;
      if (gn >= N) continue;
      const float bv = bias[gn];
#pragma unroll
      for (int e = 0; e < 8; e++) {
        const int gm = bm + wm * 32 + i * 16 + rb + e;
        float v = acc[i][j][e] + bv;
        if (OMODE == 1) {
          v = v > 0.0f ? v : 0.0f;
          ((bf16*)Cptr)[(size_t)gm * ldc + gn] = (bf16)v;
        } else {
          ((float*)Cptr)[(size_t)gm * ldc + gn] = v;
        }
      }
    }
  }
}

// ---------------------------------------------------------------------------
// Persistent LSTM scan. 128 WGs; WG `blk` owns hidden units [blk*8, blk*8+8)
// i.e. 32 gate columns (4 gates x 8 units). W_hh slice (32x1024 bf16 = 64KB)
// lives in LDS for the entire scan; cell state c lives in registers.
// Per step: gates[64,32] = h_prev[64,1024] @ Wslice^T  (8 WMMA tiles, one per
// wave), + gx, nonlinearities, h written bf16 to h_all[B,T,H]; grid barrier
// via monotonic L2 atomic counter.
// ---------------------------------------------------------------------------
__global__ __launch_bounds__(256) void lstm_scan(
    const bf16* __restrict__ Whh, const float* __restrict__ gx,
    bf16* __restrict__ h_all, int* __restrict__ syncp, int nwg)
{
  extern __shared__ char smem[];
  bf16*  Ws   = (bf16*)smem;                      // 32*1024 bf16 = 64KB
  float* gbuf = (float*)(smem + 32 * 1024 * 2);   // 64*32 fp32  =  8KB

  const int tid  = threadIdx.x;
  const int lane = tid & 31, wid = tid >> 5;
  const int blk  = blockIdx.x;                    // 0..127

  // Load W_hh slice into LDS (rows packed gate-major: lc = gate*8 + u)
  {
    const int lc   = tid >> 3;                    // 0..31
    const int gate = lc >> 3, u = lc & 7;
    const int seg  = (tid & 7) * 128;
    const uint4* s = (const uint4*)(Whh + (size_t)(gate * HID + blk * 8 + u) * HID + seg);
    uint4* d = (uint4*)&Ws[lc * HID + seg];
#pragma unroll
    for (int q = 0; q < 16; q++) d[q] = s[q];
  }
  __syncthreads();

  const int wm = wid >> 1, wn = wid & 1;          // 4 M-waves x 2 N-waves
  const int m0 = wm * 16, n0 = wn * 16;
  const int r   = lane & 15;
  const int koA = (lane >> 4) * 8;
  const int koB = (lane >> 4) * 8;

  // Per-thread cell state: 2 (batch,unit) pairs
  const int p0 = tid * 2, p1 = tid * 2 + 1;
  const int b0_ = p0 >> 3, u0_ = p0 & 7;
  const int b1_ = p1 >> 3, u1_ = p1 & 7;
  float c0 = 0.0f, c1 = 0.0f;
  const v8f zero = {};

  for (int t = 0; t < SEQ; ++t) {
    v8f acc = zero;
    if (t > 0) {
      const bf16* hrow = h_all + ((size_t)(m0 + r) * SEQ + (t - 1)) * HID + koA;
      const bf16* wrow = &Ws[(n0 + r) * HID + koB];
#pragma unroll 4
      for (int kc = 0; kc < HID; kc += 32) {
        FragU a, b;
        const uint4* pa = (const uint4*)(hrow + kc);
        a.u[0] = pa[0]; a.u[1] = pa[2];
        const uint4* pb = (const uint4*)(wrow + kc);
        b.u[0] = pb[0]; b.u[1] = pb[2];
        acc = WMMA_BF16(a.v, b.v, acc);
      }
    }
    // gates -> LDS (add precomputed gx which already contains b_ih + b_hh)
    {
      const int rb   = (lane >> 4) * 8;
      const int lc   = n0 + (lane & 15);
      const int gate = lc >> 3, uu = lc & 7;
      const int gcol = gate * HID + blk * 8 + uu;
#pragma unroll
      for (int e = 0; e < 8; e++) {
        const int bb = m0 + rb + e;
        gbuf[bb * 32 + lc] = acc[e] + gx[((size_t)bb * SEQ + t) * GATES + gcol];
      }
    }
    __syncthreads();
    // Nonlinearities + state update (PyTorch gate order i,f,g,o)
    {
      float gi = sigf(gbuf[b0_ * 32 + u0_]);
      float gf = sigf(gbuf[b0_ * 32 + 8 + u0_]);
      float gg = tanhf(gbuf[b0_ * 32 + 16 + u0_]);
      float go = sigf(gbuf[b0_ * 32 + 24 + u0_]);
      c0 = gf * c0 + gi * gg;
      h_all[((size_t)b0_ * SEQ + t) * HID + blk * 8 + u0_] = (bf16)(go * tanhf(c0));

      gi = sigf(gbuf[b1_ * 32 + u1_]);
      gf = sigf(gbuf[b1_ * 32 + 8 + u1_]);
      gg = tanhf(gbuf[b1_ * 32 + 16 + u1_]);
      go = sigf(gbuf[b1_ * 32 + 24 + u1_]);
      c1 = gf * c1 + gi * gg;
      h_all[((size_t)b1_ * SEQ + t) * HID + blk * 8 + u1_] = (bf16)(go * tanhf(c1));
    }
    __threadfence();
    __syncthreads();
    // Grid barrier: monotonic counter, no reset needed (512*128 arrivals)
    if (tid == 0) {
      __hip_atomic_fetch_add(syncp, 1, __ATOMIC_ACQ_REL, __HIP_MEMORY_SCOPE_AGENT);
      const int target = (t + 1) * nwg;
      while (__hip_atomic_load(syncp, __ATOMIC_ACQUIRE, __HIP_MEMORY_SCOPE_AGENT) < target)
        __builtin_amdgcn_s_sleep(1);
    }
    __syncthreads();
  }
}

// ---------------------------------------------------------------------------
__global__ void f2bf_kernel(const float* __restrict__ src, bf16* __restrict__ dst, int n) {
  int i = blockIdx.x * blockDim.x + threadIdx.x;
  const int stride = gridDim.x * blockDim.x;
  for (; i < n; i += stride) dst[i] = (bf16)src[i];
}

__global__ void bias2_kernel(const float* __restrict__ a, const float* __restrict__ b,
                             float* __restrict__ o, int n) {
  int i = blockIdx.x * blockDim.x + threadIdx.x;
  if (i < n) o[i] = a[i] + b[i];
}

// ---------------------------------------------------------------------------
extern "C" void kernel_launch(void* const* d_in, const int* in_sizes, int n_in,
                              void* d_out, int out_size, void* d_ws, size_t ws_size,
                              hipStream_t stream) {
  const float* feat = (const float*)d_in[0];
  const float* Wpre = (const float*)d_in[1];
  const float* bpre = (const float*)d_in[2];
  const float* Wih  = (const float*)d_in[3];
  const float* bih  = (const float*)d_in[4];
  const float* Whh  = (const float*)d_in[5];
  const float* bhh  = (const float*)d_in[6];
  const float* Wcls = (const float*)d_in[7];
  const float* bcls = (const float*)d_in[8];
  float* out = (float*)d_out;

  char* ws = (char*)d_ws;
  size_t off = 0;
  auto carve = [&](size_t bytes) { size_t o = off; off = (off + bytes + 255) & ~(size_t)255; return o; };
  bf16*  Wpre_b = (bf16*)(ws + carve((size_t)HID * FEAT * 2));
  bf16*  Wih_b  = (bf16*)(ws + carve((size_t)GATES * HID * 2));
  bf16*  Whh_b  = (bf16*)(ws + carve((size_t)GATES * HID * 2));
  bf16*  Wcls_b = (bf16*)(ws + carve((size_t)NCLS * HID * 2));
  float* bias2  = (float*)(ws + carve((size_t)GATES * 4));
  bf16*  x_b    = (bf16*)(ws + carve((size_t)MROWS * HID * 2));
  float* gx     = (float*)(ws + carve((size_t)MROWS * GATES * 4));
  bf16*  h_all  = (bf16*)(ws + carve((size_t)MROWS * HID * 2));
  int*   syncp  = (int*)(ws + carve(256));

  hipMemsetAsync(syncp, 0, sizeof(int), stream);

  f2bf_kernel<<<1024, 256, 0, stream>>>(Wpre, Wpre_b, HID * FEAT);
  f2bf_kernel<<<1024, 256, 0, stream>>>(Wih,  Wih_b,  GATES * HID);
  f2bf_kernel<<<1024, 256, 0, stream>>>(Whh,  Whh_b,  GATES * HID);
  f2bf_kernel<<<96,   256, 0, stream>>>(Wcls, Wcls_b, NCLS * HID);
  bias2_kernel<<<16,  256, 0, stream>>>(bih, bhh, bias2, GATES);

  // x = relu(feat @ W_pre^T + b_pre)  -> bf16
  gemm_bf16_wmma<0, 1><<<dim3(MROWS / 128, HID / 128), dim3(256), 0, stream>>>(
      feat, Wpre_b, bpre, x_b, MROWS, HID, FEAT, HID);
  // gx = x @ W_ih^T + (b_ih + b_hh)   -> fp32
  gemm_bf16_wmma<1, 0><<<dim3(MROWS / 128, GATES / 128), dim3(256), 0, stream>>>(
      x_b, Wih_b, bias2, gx, MROWS, GATES, HID, GATES);
  // recurrent scan -> h_all (bf16, [B,T,H])
  lstm_scan<<<dim3(128), dim3(256), 32 * 1024 * 2 + 64 * 32 * 4, stream>>>(
      Whh_b, gx, h_all, syncp, 128);
  // scores = h_all @ W_cls^T + b_cls  -> fp32 (rows b*T+t match output layout)
  gemm_bf16_wmma<1, 0><<<dim3(MROWS / 128, 1), dim3(256), 0, stream>>>(
      h_all, Wcls_b, bcls, out, MROWS, NCLS, HID, NCLS);
}
